// SamVisionAttention_25658134626437
// MI455X (gfx1250) — matrix operand order repaired and checked by reading, not compiled, not run on hardware
//
#include <hip/hip_runtime.h>
#include <hip/hip_bf16.h>

typedef __attribute__((ext_vector_type(16))) __bf16 v16bf;
typedef __attribute__((ext_vector_type(8)))  float  v8f;
typedef int v4i_g __attribute__((vector_size(16)));   // matches builtin pointee

#define NH   12
#define HD   64
#define HW   196
#define MT   13      // 13 query tiles (208/16)
#define NT   14      // 14 key tiles (224/16)
#define KP   224     // padded key count
#define PS_STR 228   // P buffer row stride (bf16 elems), padded vs 224 for banks
#define WT_STR 132   // W-tile row stride (bf16 elems), padded vs 128 for banks
#define BT_STR 36    // rel-table B row stride (bf16 elems), padded vs 32 for banks

#if __has_builtin(__builtin_amdgcn_global_load_async_to_lds_b128)
#define HAVE_ASYNC_LDS 1
#endif

#ifdef HAVE_ASYNC_LDS
#define GPTR(p) ((__attribute__((address_space(1))) v4i_g*)(uintptr_t)(p))
#define LPTR(p) ((__attribute__((address_space(3))) v4i_g*)(uintptr_t)(p))
#endif

// Assemble a v16bf WMMA fragment from two 16-byte halves (8-byte aligned).
static __device__ __forceinline__ v16bf load16(const __bf16* p0, const __bf16* p1) {
  union { v16bf v; unsigned long long u[4]; } r;
  const unsigned long long* a = (const unsigned long long*)p0;
  const unsigned long long* b = (const unsigned long long*)p1;
  r.u[0] = a[0]; r.u[1] = a[1]; r.u[2] = b[0]; r.u[3] = b[1];
  return r.v;
}

// ---------------------------------------------------------------- zero fill
__global__ void zero_kernel(float4* __restrict__ p, long long n4) {
  long long i = (long long)blockIdx.x * blockDim.x + threadIdx.x;
  long long stride = (long long)gridDim.x * blockDim.x;
  float4 z = {0.f, 0.f, 0.f, 0.f};
  for (; i < n4; i += stride) p[i] = z;
}

// ---------------------------------------------------------------- QKV GEMM
// X[50176,768] fp32 @ Wq^T[768,2304] + b -> scatter into q(scaled)/kT/v bf16.
__global__ __launch_bounds__(256) void qkv_kernel(
    const float* __restrict__ X, const float* __restrict__ Wq,
    const float* __restrict__ bias, __bf16* __restrict__ q_s,
    __bf16* __restrict__ kT_s, __bf16* __restrict__ v_s) {
  __shared__ __attribute__((aligned(16))) __bf16 Wt[32 * WT_STR];

  const int t = threadIdx.x;
  const int lane = t & 31, w = t >> 5;
  const int wr = w >> 2, wc = w & 3;          // 2 x 4 wave grid
  const int lm = lane & 15, hl = lane >> 4;
  const int nb0 = blockIdx.x * 128;
  const int mb0 = blockIdx.y * 64;

  v8f acc[2][2] = {};

  for (int kb = 0; kb < 768; kb += 32) {
    // Stage W[nb0..nb0+127][kb..kb+31] transposed -> Wt[k][n] (bf16)
    #pragma unroll
    for (int i = 0; i < 4; ++i) {
      int task = t + i * 256;               // 1024 float4 tasks
      int n = task >> 3, kc = task & 7;
      const float4 f = *(const float4*)(Wq + (size_t)(nb0 + n) * 768 + kb + kc * 4);
      int k0 = kc * 4;
      Wt[(k0 + 0) * WT_STR + n] = (__bf16)f.x;
      Wt[(k0 + 1) * WT_STR + n] = (__bf16)f.y;
      Wt[(k0 + 2) * WT_STR + n] = (__bf16)f.z;
      Wt[(k0 + 3) * WT_STR + n] = (__bf16)f.w;
    }
    __syncthreads();

    // A fragments direct from global fp32 (lane = M row, K halves 0-7 / 16-23)
    v16bf a[2];
    #pragma unroll
    for (int sm = 0; sm < 2; ++sm) {
      int m = mb0 + wr * 32 + sm * 16 + lm;
      const float* p = X + (size_t)m * 768 + kb + hl * 8;
      float fb[16];
      *(float4*)&fb[0]  = *(const float4*)(p);
      *(float4*)&fb[4]  = *(const float4*)(p + 4);
      *(float4*)&fb[8]  = *(const float4*)(p + 16);
      *(float4*)&fb[12] = *(const float4*)(p + 20);
      v16bf av;
      #pragma unroll
      for (int j = 0; j < 16; ++j) av[j] = (__bf16)fb[j];
      a[sm] = av;
    }
    // B fragments from LDS (lane = N column, K strided rows 0-7 / 16-23)
    #pragma unroll
    for (int sn = 0; sn < 2; ++sn) {
      const __bf16* pb = &Wt[(hl * 8) * WT_STR + wc * 32 + sn * 16 + lm];
      v16bf b;
      for (int s = 0; s < 16; ++s) b[s] = pb[((s & 7) + (s >> 3) * 16) * WT_STR];
      #pragma unroll
      for (int sm = 0; sm < 2; ++sm)
        acc[sm][sn] = __builtin_amdgcn_wmma_f32_16x16x32_bf16(
            false, a[sm], false, b, (short)0, acc[sm][sn], false, false);
    }
    __syncthreads();
  }

  // Epilogue: bias, scatter into q (scaled) / kT (dim-major, padded) / v
  #pragma unroll
  for (int sm = 0; sm < 2; ++sm)
    #pragma unroll
    for (int sn = 0; sn < 2; ++sn)
      #pragma unroll
      for (int r = 0; r < 8; ++r) {
        int m = mb0 + wr * 32 + sm * 16 + r + 8 * hl;
        int n = nb0 + wc * 32 + sn * 16 + lm;
        float val = acc[sm][sn][r] + bias[n];
        int which = n / 768, rem = n % 768;
        int head = rem >> 6, d = rem & 63;
        size_t b3 = (size_t)(m / HW) * NH + head;
        int p = m % HW;
        if (which == 0)
          q_s[(b3 * HW + p) * HD + d] = (__bf16)(val * 0.125f);   // * hd^-0.5
        else if (which == 1)
          kT_s[(b3 * HD + d) * KP + p] = (__bf16)val;
        else
          v_s[(b3 * HW + p) * HD + d] = (__bf16)val;
      }
}

// ---------------------------------------------------------------- Rel GEMM
// Th[m, j] = 8 * dot(q_scaled[m,:], rel_pos_h[j,:])  for all 602112 token rows,
// j in 0..26 (padded to 32).  Same for Tw.  M = 602112 = 37632 x 16 exactly.
__global__ __launch_bounds__(256) void relgemm_kernel(
    const __bf16* __restrict__ q_s, const float* __restrict__ rph,
    const float* __restrict__ rpw, __bf16* __restrict__ Th,
    __bf16* __restrict__ Tw) {
  __shared__ __attribute__((aligned(16))) __bf16 Bt[2][64 * BT_STR];

  const int t = threadIdx.x;
  // Stage both 27x64 tables transposed -> Bt[tensor][d][j], zero pad j=27..31
  for (int i = t; i < 2 * 64 * 32; i += 256) {
    int which = i >> 11;                 // 64*32 = 2048 per tensor
    int rem = i & 2047;
    int d = rem >> 5, j = rem & 31;
    const float* src = which ? rpw : rph;
    float v = (j < 27) ? src[j * 64 + d] : 0.f;
    Bt[which][d * BT_STR + j] = (__bf16)v;
  }
  __syncthreads();

  const int lane = t & 31, w = t >> 5;
  const int lm = lane & 15, hl = lane >> 4;
  const size_t tile = (size_t)blockIdx.x * 8 + w;   // < 37632
  const size_t mrow = tile * 16 + lm;

  v8f acc[2][2] = {};                     // [tensor][ntile]
  #pragma unroll
  for (int ks = 0; ks < HD; ks += 32) {
    const __bf16* pa = q_s + mrow * HD + ks + hl * 8;
    v16bf a = load16(pa, pa + 16);
    #pragma unroll
    for (int tens = 0; tens < 2; ++tens)
      #pragma unroll
      for (int tn = 0; tn < 2; ++tn) {
        const __bf16* pb = &Bt[tens][(ks + hl * 8) * BT_STR + tn * 16 + lm];
        v16bf b;
        for (int s = 0; s < 16; ++s) b[s] = pb[((s & 7) + (s >> 3) * 16) * BT_STR];
        acc[tens][tn] = __builtin_amdgcn_wmma_f32_16x16x32_bf16(
            false, a, false, b, (short)0, acc[tens][tn], false, false);
      }
  }

  const size_t mbase = tile * 16 * 32;
  #pragma unroll
  for (int r = 0; r < 8; ++r) {
    int mloc = r + 8 * hl;
    #pragma unroll
    for (int tn = 0; tn < 2; ++tn) {
      size_t off = mbase + (size_t)mloc * 32 + tn * 16 + lm;
      Th[off] = (__bf16)(acc[0][tn][r] * 8.f);   // undo q pre-scale
      Tw[off] = (__bf16)(acc[1][tn][r] * 8.f);
    }
  }
}

// ---------------------------------------------------------------- Attention
__global__ __launch_bounds__(128) void attn_kernel(
    const __bf16* __restrict__ q_s, const __bf16* __restrict__ kT_s,
    const __bf16* __restrict__ v_s, const __bf16* __restrict__ Th,
    const __bf16* __restrict__ Tw, __bf16* __restrict__ ao) {
  __shared__ __attribute__((aligned(16))) __bf16 Vs[KP * HD];          // 28672 B
  __shared__ __attribute__((aligned(16))) __bf16 Ps[4][16 * PS_STR];   // 29184 B
  __shared__ __attribute__((aligned(16))) float relH[4][16][14];       // 3584 B
  __shared__ __attribute__((aligned(16))) float relW[4][16][14];       // 3584 B

  const int bi = blockIdx.x;
  const int t = threadIdx.x;
  const int lane = t & 31, wv = t >> 5;     // 4 waves
  const int lm = lane & 15, hl = lane >> 4;
  const size_t bq = (size_t)bi * HW * HD;   // q / v base (elements)
  const size_t bk = (size_t)bi * HD * KP;   // kT base
  const size_t b32 = (size_t)bi * HW * 32;  // Th/Tw base
  const int bimg = bi / NH, head = bi % NH;

  // Cooperative V -> LDS (pad rows zeroed)
#ifdef HAVE_ASYNC_LDS
  {
    const __bf16* gsrc = v_s + bq;
    // rows 0..195: async global -> LDS, 16B per lane, 1568 chunks
    for (int i = t; i < (HW * HD) / 8; i += 128)
      __builtin_amdgcn_global_load_async_to_lds_b128(
          GPTR(gsrc + i * 8), LPTR(Vs + i * 8), 0, 0);
    // rows 196..223: zero fill (224 x 16B)
    for (int i = t; i < ((KP - HW) * HD) / 8; i += 128)
      *(uint4*)((char*)Vs + HW * HD * 2 + i * 16) = make_uint4(0u, 0u, 0u, 0u);
#if __has_builtin(__builtin_amdgcn_s_wait_asynccnt)
    __builtin_amdgcn_s_wait_asynccnt(0);
#else
    asm volatile("s_wait_asynccnt 0" ::: "memory");
#endif
  }
#else
  {
    const unsigned int* vsrc = (const unsigned int*)(v_s + bq);
    unsigned int* vdst = (unsigned int*)Vs;
    for (int i = t; i < KP * HD / 2; i += 128) {
      int row = i >> 5;                     // 32 uints (64 bf16) per row
      vdst[i] = (row < HW) ? vsrc[i] : 0u;
    }
  }
#endif
  __syncthreads();

  for (int tm = wv; tm < MT; tm += 4) {
    // Gather per-wave rel-pos tables from precomputed Th/Tw
    for (int task = lane; task < 224; task += 32) {      // 16 rows x 14 keys
      int r = task / 14, kk = task % 14;
      int m = tm * 16 + r;
      float sh = 0.f, sw = 0.f;
      if (m < HW) {
        int h = m / 14, ww = m % 14;
        sh = (float)Th[b32 + (size_t)m * 32 + (h - kk + 13)];
        sw = (float)Tw[b32 + (size_t)m * 32 + (ww - kk + 13)];
      }
      relH[wv][r][kk] = sh;
      relW[wv][r][kk] = sw;
    }
    asm volatile("s_wait_dscnt 0" ::: "memory");

    // S = (q*scale) @ k^T  -- 14 N-tiles, K=64 in two steps of 32
    v8f acc[NT] = {};
    for (int ks = 0; ks < HD; ks += 32) {
      int mrow = tm * 16 + lm; if (mrow > HW - 1) mrow = HW - 1;  // clamp pad rows
      const __bf16* pa = q_s + bq + (size_t)mrow * HD + ks + hl * 8;
      v16bf a = load16(pa, pa + 16);
      #pragma unroll
      for (int tn = 0; tn < NT; ++tn) {
        const __bf16* pb = kT_s + bk + (size_t)(ks + hl * 8) * KP + tn * 16 + lm;
        v16bf b;
        for (int s = 0; s < 16; ++s) b[s] = pb[((s & 7) + (s >> 3) * 16) * KP];
        acc[tn] = __builtin_amdgcn_wmma_f32_16x16x32_bf16(
            false, a, false, b, (short)0, acc[tn], false, false);
      }
    }

    // Add rel-pos, mask padded keys / neutralize padded rows
    #pragma unroll
    for (int tn = 0; tn < NT; ++tn) {
      int key = tn * 16 + lm;
      int kh = key / 14, kw = key % 14;
      #pragma unroll
      for (int r = 0; r < 8; ++r) {
        int mloc = r + 8 * hl;
        int m = tm * 16 + mloc;
        float s = acc[tn][r];
        if (m >= HW)       s = 0.f;
        else if (key < HW) s += relH[wv][mloc][kh] + relW[wv][mloc][kw];
        else               s = -3.0e38f;
        acc[tn][r] = s;
      }
    }

    // Row-wise softmax (reduce across 16-lane halves with shfl_xor)
    float mx[8], sm[8], inv[8];
    #pragma unroll
    for (int r = 0; r < 8; ++r) {
      float m0 = acc[0][r];
      #pragma unroll
      for (int tn = 1; tn < NT; ++tn) m0 = fmaxf(m0, acc[tn][r]);
      m0 = fmaxf(m0, __shfl_xor(m0, 1, 32));
      m0 = fmaxf(m0, __shfl_xor(m0, 2, 32));
      m0 = fmaxf(m0, __shfl_xor(m0, 4, 32));
      m0 = fmaxf(m0, __shfl_xor(m0, 8, 32));
      mx[r] = m0;
      sm[r] = 0.f;
    }
    #pragma unroll
    for (int tn = 0; tn < NT; ++tn)
      #pragma unroll
      for (int r = 0; r < 8; ++r) {
        int mloc = r + 8 * hl;
        float p = __expf(acc[tn][r] - mx[r]);
        sm[r] += p;
        Ps[wv][mloc * PS_STR + tn * 16 + lm] = (__bf16)p;   // C->A relayout via LDS
      }
    #pragma unroll
    for (int r = 0; r < 8; ++r) {
      float s0 = sm[r];
      s0 += __shfl_xor(s0, 1, 32);
      s0 += __shfl_xor(s0, 2, 32);
      s0 += __shfl_xor(s0, 4, 32);
      s0 += __shfl_xor(s0, 8, 32);
      inv[r] = 1.f / s0;
    }
    asm volatile("s_wait_dscnt 0" ::: "memory");

    // O = P @ V : K = 224 keys in 7 steps of 32, N = 64 dims (4 tiles)
    v8f o[4] = {};
    for (int kb = 0; kb < KP; kb += 32) {
      const __bf16* pa = &Ps[wv][lm * PS_STR + kb + hl * 8];
      v16bf a = load16(pa, pa + 16);
      #pragma unroll
      for (int tn = 0; tn < 4; ++tn) {
        const __bf16* pb = &Vs[(kb + hl * 8) * HD + tn * 16 + lm];
        v16bf b;
        for (int s = 0; s < 16; ++s) b[s] = pb[((s & 7) + (s >> 3) * 16) * HD];
        o[tn] = __builtin_amdgcn_wmma_f32_16x16x32_bf16(
            false, a, false, b, (short)0, o[tn], false, false);
      }
    }

    // Normalize and store into [B, HW, C] bf16 layout for proj
    #pragma unroll
    for (int r = 0; r < 8; ++r) {
      int mloc = r + 8 * hl;
      int m = tm * 16 + mloc;
      if (m < HW) {
        size_t base = ((size_t)bimg * HW + m) * 768 + head * HD + lm;
        #pragma unroll
        for (int tn = 0; tn < 4; ++tn)
          ao[base + tn * 16] = (__bf16)(o[tn][r] * inv[r]);
      }
    }
  }
}

// ---------------------------------------------------------------- Proj GEMM
// A[50176,768] bf16 @ Wp^T[768,768] + b -> out fp32
__global__ __launch_bounds__(256) void proj_kernel(
    const __bf16* __restrict__ A, const float* __restrict__ Wp,
    const float* __restrict__ bias, float* __restrict__ out) {
  __shared__ __attribute__((aligned(16))) __bf16 Wt[32 * WT_STR];

  const int t = threadIdx.x;
  const int lane = t & 31, w = t >> 5;
  const int wr = w >> 2, wc = w & 3;
  const int lm = lane & 15, hl = lane >> 4;
  const int nb0 = blockIdx.x * 128;
  const int mb0 = blockIdx.y * 64;

  v8f acc[2][2] = {};

  for (int kb = 0; kb < 768; kb += 32) {
    #pragma unroll
    for (int i = 0; i < 4; ++i) {
      int task = t + i * 256;
      int n = task >> 3, kc = task & 7;
      const float4 f = *(const float4*)(Wp + (size_t)(nb0 + n) * 768 + kb + kc * 4);
      int k0 = kc * 4;
      Wt[(k0 + 0) * WT_STR + n] = (__bf16)f.x;
      Wt[(k0 + 1) * WT_STR + n] = (__bf16)f.y;
      Wt[(k0 + 2) * WT_STR + n] = (__bf16)f.z;
      Wt[(k0 + 3) * WT_STR + n] = (__bf16)f.w;
    }
    __syncthreads();

    v16bf a[2];
    #pragma unroll
    for (int sm = 0; sm < 2; ++sm) {
      int m = mb0 + wr * 32 + sm * 16 + lm;
      const __bf16* p = A + (size_t)m * 768 + kb + hl * 8;
      a[sm] = load16(p, p + 16);
    }
    #pragma unroll
    for (int sn = 0; sn < 2; ++sn) {
      const __bf16* pb = &Wt[(hl * 8) * WT_STR + wc * 32 + sn * 16 + lm];
      v16bf b;
      for (int s = 0; s < 16; ++s) b[s] = pb[((s & 7) + (s >> 3) * 16) * WT_STR];
      #pragma unroll
      for (int sm = 0; sm < 2; ++sm)
        acc[sm][sn] = __builtin_amdgcn_wmma_f32_16x16x32_bf16(
            false, a[sm], false, b, (short)0, acc[sm][sn], false, false);
    }
    __syncthreads();
  }

  #pragma unroll
  for (int sm = 0; sm < 2; ++sm)
    #pragma unroll
    for (int sn = 0; sn < 2; ++sn)
      #pragma unroll
      for (int r = 0; r < 8; ++r) {
        int m = mb0 + wr * 32 + sm * 16 + r + 8 * hl;
        int n = nb0 + wc * 32 + sn * 16 + lm;
        out[(size_t)m * 768 + n] = acc[sm][sn][r] + bias[n];
      }
}

// ---------------------------------------------------------------- launcher
extern "C" void kernel_launch(void* const* d_in, const int* in_sizes, int n_in,
                              void* d_out, int out_size, void* d_ws, size_t ws_size,
                              hipStream_t stream) {
  (void)in_sizes; (void)n_in; (void)out_size; (void)ws_size;
  const float* X    = (const float*)d_in[0];  // hidden_states [256,14,14,768]
  const float* qkvw = (const float*)d_in[1];  // [2304,768]
  const float* qkvb = (const float*)d_in[2];  // [2304]
  const float* rph  = (const float*)d_in[3];  // [27,64]
  const float* rpw  = (const float*)d_in[4];  // [27,64]
  const float* pw   = (const float*)d_in[5];  // [768,768]
  const float* pb   = (const float*)d_in[6];  // [768]
  float* out = (float*)d_out;

  char* ws = (char*)d_ws;
  __bf16* q_s  = (__bf16*)ws;  ws += (size_t)3072 * HW * HD * 2;  // 77,070,336 B
  __bf16* kT_s = (__bf16*)ws;  ws += (size_t)3072 * HD * KP * 2;  // 88,080,384 B
  __bf16* v_s  = (__bf16*)ws;  ws += (size_t)3072 * HW * HD * 2;  // 77,070,336 B
  __bf16* ao   = (__bf16*)ws;  ws += (size_t)3072 * HW * HD * 2;  // 77,070,336 B
  __bf16* Th   = (__bf16*)ws;  ws += (size_t)3072 * HW * 32 * 2;  // 38,535,168 B
  __bf16* Tw   = (__bf16*)ws;                                     // 38,535,168 B

  long long n4 = ((long long)3072 * HD * KP * 2) / 16;
  zero_kernel<<<2048, 256, 0, stream>>>((float4*)kT_s, n4);
  qkv_kernel<<<dim3(18, 784), 256, 0, stream>>>(X, qkvw, qkvb, q_s, kT_s, v_s);
  relgemm_kernel<<<4704, 256, 0, stream>>>(q_s, rph, rpw, Th, Tw);
  attn_kernel<<<3072, 128, 0, stream>>>(q_s, kT_s, v_s, Th, Tw, ao);
  proj_kernel<<<dim3(6, 784), 256, 0, stream>>>(ao, pw, pb, out);
}
